// MultiHeadAttention_326417514651
// MI455X (gfx1250) — matrix operand order repaired and checked by reading, not compiled, not run on hardware
//
#include <hip/hip_runtime.h>

// ---------------- problem constants ----------------
#define B_   4
#define S_   2048
#define H_   16
#define DH_  64
#define D_   1024
#define M_   (B_ * S_)          // 8192 rows total
#define NK_  (D_ / 32)          // k-steps
#define SCALE_ 0.125f           // 1/sqrt(64)
#define LOG2E_ 1.44269504088896340736f

typedef __attribute__((ext_vector_type(16))) __bf16 v16bf;
typedef __attribute__((ext_vector_type(8)))  float  v8f;
typedef __attribute__((ext_vector_type(4)))  unsigned int u32x4;
typedef __attribute__((ext_vector_type(8)))  int    i32x8;
typedef __attribute__((ext_vector_type(4)))  int    i32x4;

union Frag16 { v16bf v; uint4 q[2]; };

__device__ __forceinline__ unsigned short f2bf(float f) {
    unsigned int u = __float_as_uint(f);
    u += 0x7FFFu + ((u >> 16) & 1u);   // round-to-nearest-even
    return (unsigned short)(u >> 16);
}

__device__ __forceinline__ v8f wmma_bf16(v16bf a, v16bf b, v8f c) {
    // D = A(16x32 bf16) * B(32x16 bf16) + C(16x16 f32)
    return __builtin_amdgcn_wmma_f32_16x16x32_bf16(false, a, false, b,
                                                   (short)0, c, false, false);
}

// ---- TDM: async DMA of a 64x32 bf16 tile (row stride D_) into LDS ----
// D# built per CDNA5 ISA. data_size=1 (2 bytes),
// tensor 1024x1024, tile dim0=32 (k, contiguous), dim1=64 (cols), stride=1024.
__device__ __forceinline__ void tdm_load_tile_64x32(unsigned lds_off,
                                                    const unsigned short* gptr) {
    unsigned long long ga = (unsigned long long)(uintptr_t)gptr;
    u32x4 g0;
    g0[0] = 1u;                                       // count=1, user descriptor
    g0[1] = lds_off;                                  // lds_addr (bytes)
    g0[2] = (unsigned)(ga & 0xFFFFFFFFull);           // global_addr[31:0]
    g0[3] = ((unsigned)((ga >> 32) & 0x01FFFFFFull))  // global_addr[56:32]
            | 0x80000000u;                            // type=2 ("image")
    i32x8 g1;
    g1[0] = 0x00010000;            // [17:16] data_size=1 -> 2 bytes
    g1[1] = (int)(1024u << 16);    // [63:48]  tensor_dim0 lo16 = 1024
    g1[2] = (int)(1024u << 16);    // [79:64]  dim0 hi=0 | [95:80] tensor_dim1 lo16 = 1024
    g1[3] = (int)(32u << 16);      // [111:96] dim1 hi=0 | [127:112] tile_dim0 = 32
    g1[4] = 64;                    // [143:128] tile_dim1 = 64 | tile_dim2 = 0
    g1[5] = 1024;                  // [191:160] tensor_dim0_stride lo32 = 1024
    g1[6] = 0;                     // stride hi | tensor_dim1_stride lo (2D: unused)
    g1[7] = 0;
    i32x4 z4 = {0, 0, 0, 0};       // groups 2/3 unused (2D tensor)
    i32x8 z8 = {0, 0, 0, 0, 0, 0, 0, 0};
    __builtin_amdgcn_tensor_load_to_lds(g0, g1, z4, z4, z8, 0);
}

// ---------------- 1) f32 -> bf16 elementwise ----------------
__global__ void __launch_bounds__(256)
cvt_f32_bf16_kernel(const float* __restrict__ in, unsigned short* __restrict__ out, int n) {
    int i = (blockIdx.x * 256 + threadIdx.x) * 4;
    if (i < n) {
        float4 v = *(const float4*)(in + i);
        unsigned int u0 = (unsigned int)f2bf(v.x) | ((unsigned int)f2bf(v.y) << 16);
        unsigned int u1 = (unsigned int)f2bf(v.z) | ((unsigned int)f2bf(v.w) << 16);
        uint2 o; o.x = u0; o.y = u1;
        *(uint2*)(out + i) = o;
    }
}

// ---------------- 2) f32 W[D,D] -> bf16 W^T[D,D] (LDS tiled) ----------------
__global__ void __launch_bounds__(256)
transpose_cvt_kernel(const float* __restrict__ W, unsigned short* __restrict__ WT) {
    __shared__ unsigned short tile[32][33];
    const int bx = blockIdx.x * 32;   // column tile of W
    const int by = blockIdx.y * 32;   // row tile of W
    const int tx = threadIdx.x & 31;
    const int ty = threadIdx.x >> 5;  // 0..7
#pragma unroll
    for (int j = 0; j < 32; j += 8)
        tile[ty + j][tx] = f2bf(W[(size_t)(by + ty + j) * D_ + (bx + tx)]);
    __syncthreads();
#pragma unroll
    for (int j = 0; j < 32; j += 8)
        WT[(size_t)(bx + ty + j) * D_ + (by + tx)] = tile[tx][ty + j];
}

// ---------------- 3) WMMA GEMM with TDM-staged B tiles ----------------
// C[M,N] = A[M,K=D] * BT[N,K=D]^T + bias
// Block: 8 waves, each wave computes 32x64 (2 m-tiles x 4 n-tiles).
// B tile (64 cols x 32 k, 4 KB) DMA'd into LDS by the TDM, double-buffered.
// MODE 0: bf16 out row-major [M,D]; MODE 1: bf16 scattered to Vt [B,H,dh,S];
// MODE 2: f32 out row-major [M,D]
template <int MODE>
__global__ void __launch_bounds__(256)
gemm_bf16_kernel(const unsigned short* __restrict__ A,
                 const unsigned short* __restrict__ BT,
                 const float* __restrict__ bias,
                 void* __restrict__ Out) {
    __shared__ unsigned short bstage[2][64 * 32];   // 2 x 4 KB double buffer

    const int wave = threadIdx.x >> 5;
    const int lane = threadIdx.x & 31;
    const int half = lane >> 4;
    const int ln   = lane & 15;

    const int m_base = blockIdx.x * 256 + wave * 32;   // 8 waves x 32 rows
    const int n_base = blockIdx.y * 64;                // 4 WMMA n-tiles (shared via LDS)

    v8f zero = {0.f,0.f,0.f,0.f,0.f,0.f,0.f,0.f};
    v8f acc[2][4] = {{zero, zero, zero, zero}, {zero, zero, zero, zero}};

    const unsigned short* Arow0 = A + (size_t)(m_base + ln) * D_;
    const unsigned short* Arow1 = A + (size_t)(m_base + 16 + ln) * D_;
    const unsigned short* Btile = BT + (size_t)n_base * D_;   // [64 rows][D k] bf16
    const unsigned lds_base = (unsigned)(uintptr_t)(&bstage[0][0]);

    if (wave == 0) tdm_load_tile_64x32(lds_base, Btile);      // prologue: tile 0

    for (int i = 0; i < NK_; i++) {
        const int k0 = i * 32;
        if (wave == 0) {
            if (i + 1 < NK_) {
                tdm_load_tile_64x32(lds_base + ((unsigned)((i + 1) & 1)) * (64 * 32 * 2),
                                    Btile + (size_t)(i + 1) * 32);
                __builtin_amdgcn_s_wait_tensorcnt(1);   // tile i landed, tile i+1 in flight
            } else {
                __builtin_amdgcn_s_wait_tensorcnt(0);
            }
        }
        __syncthreads();   // B tile i visible to all 8 waves

        Frag16 af0, af1;   // A 16x32: lane row = ln; k = k0 + 8*half + [0..7], +16
        af0.q[0] = *(const uint4*)(Arow0 + k0 + half * 8);
        af0.q[1] = *(const uint4*)(Arow0 + k0 + 16 + half * 8);
        af1.q[0] = *(const uint4*)(Arow1 + k0 + half * 8);
        af1.q[1] = *(const uint4*)(Arow1 + k0 + 16 + half * 8);

        const unsigned short* Bl = &bstage[i & 1][0];
#pragma unroll
        for (int nt = 0; nt < 4; nt++) {
            // B 32x16: lane col n = nt*16+ln; elems k = 8*half + [0..7], +16
            const unsigned short* bp = Bl + (nt * 16 + ln) * 32 + half * 8;
            Frag16 bf;
            bf.q[0] = *(const uint4*)(bp);
            bf.q[1] = *(const uint4*)(bp + 16);
            acc[0][nt] = wmma_bf16(af0.v, bf.v, acc[0][nt]);
            acc[1][nt] = wmma_bf16(af1.v, bf.v, acc[1][nt]);
        }
        __syncthreads();   // done consuming before buffer i&1 is re-filled (tile i+2) at iter i+1
    }

#pragma unroll
    for (int mt = 0; mt < 2; mt++) {
#pragma unroll
        for (int nt = 0; nt < 4; nt++) {
            const int n = n_base + nt * 16 + ln;
            const float bn = bias[n];
#pragma unroll
            for (int r = 0; r < 8; r++) {
                const int m = m_base + mt * 16 + r + 8 * half;  // C/D: row = r + 8*(lane/16)
                const float o = acc[mt][nt][r] + bn;
                if (MODE == 0) {
                    ((unsigned short*)Out)[(size_t)m * D_ + n] = f2bf(o);
                } else if (MODE == 1) {
                    const int b = m / S_, s = m % S_;
                    const int h = n >> 6, d = n & 63;
                    ((unsigned short*)Out)[((size_t)(b * H_ + h) * DH_ + d) * S_ + s] = f2bf(o);
                } else {
                    ((float*)Out)[(size_t)m * D_ + n] = o;
                }
            }
        }
    }
}

// ---------------- 4) flash attention (one wave per 16-row q-tile) ----------------
__global__ void __launch_bounds__(256)
attn_kernel(const unsigned short* __restrict__ Qb,   // [B*S, D] bf16
            const unsigned short* __restrict__ Kb,   // [B*S, D] bf16
            const unsigned short* __restrict__ Vt,   // [B*H*dh, S] bf16
            unsigned short* __restrict__ Ob,         // [B*S, D] bf16
            const int* __restrict__ causal_flag) {
    __shared__ unsigned short ldsP[8 * 16 * 32];     // per-wave 16x32 P staging

    const int wave = threadIdx.x >> 5;
    const int lane = threadIdx.x & 31;
    const int half = lane >> 4;
    const int ln   = lane & 15;
    unsigned short* Pl = ldsP + wave * 16 * 32;

    const int tiles_per_bh = S_ / 128;               // 8 waves x 16 rows per block
    const int bh   = blockIdx.x / tiles_per_bh;
    const int qblk = blockIdx.x % tiles_per_bh;
    const int b = bh / H_, h = bh % H_;
    const int q_base = qblk * 128 + wave * 16;
    const bool causal = (causal_flag[0] != 0);

    // Q A-fragments (16 x 64 split into two 16x32 k-chunks), loaded once
    Frag16 qa[2];
#pragma unroll
    for (int c = 0; c < 2; c++) {
        const unsigned short* p =
            Qb + (size_t)(b * S_ + q_base + ln) * D_ + h * DH_ + c * 32 + half * 8;
        qa[c].q[0] = *(const uint4*)(p);
        qa[c].q[1] = *(const uint4*)(p + 16);
    }

    v8f zero = {0.f,0.f,0.f,0.f,0.f,0.f,0.f,0.f};
    v8f oacc[4] = {zero, zero, zero, zero};
    float mrow[8], lrow[8];
#pragma unroll
    for (int r = 0; r < 8; r++) { mrow[r] = -__builtin_inff(); lrow[r] = 0.f; }

    const float sc = SCALE_ * LOG2E_;
    const int kv_end = causal ? (q_base + 16) : S_;

    for (int kv0 = 0; kv0 < kv_end; kv0 += 32) {
        // ---- scores: two 16x16 tiles over dh=64 (two k-chunks each) ----
        v8f s0 = zero, s1 = zero;
#pragma unroll
        for (int c = 0; c < 2; c++) {
            const unsigned short* kp0 =
                Kb + (size_t)(b * S_ + kv0 + ln) * D_ + h * DH_ + c * 32 + half * 8;
            const unsigned short* kp1 =
                Kb + (size_t)(b * S_ + kv0 + 16 + ln) * D_ + h * DH_ + c * 32 + half * 8;
            Frag16 kb0, kb1;
            kb0.q[0] = *(const uint4*)(kp0);  kb0.q[1] = *(const uint4*)(kp0 + 16);
            kb1.q[0] = *(const uint4*)(kp1);  kb1.q[1] = *(const uint4*)(kp1 + 16);
            s0 = wmma_bf16(qa[c].v, kb0.v, s0);
            s1 = wmma_bf16(qa[c].v, kb1.v, s1);
        }

        // ---- online softmax (per-row over half-wave) + stage P into LDS ----
#pragma unroll
        for (int r = 0; r < 8; r++) {
            const int qg = q_base + r + 8 * half;
            float a0 = (!causal || (kv0 + ln      <= qg)) ? s0[r] * sc : -__builtin_inff();
            float a1 = (!causal || (kv0 + 16 + ln <= qg)) ? s1[r] * sc : -__builtin_inff();
            float rm = fmaxf(a0, a1);
#pragma unroll
            for (int msk = 1; msk < 16; msk <<= 1)
                rm = fmaxf(rm, __shfl_xor(rm, msk, 32));
            const float newm  = fmaxf(mrow[r], rm);
            const float alpha = exp2f(mrow[r] - newm);
            mrow[r] = newm;
            const float p0 = exp2f(a0 - newm);
            const float p1 = exp2f(a1 - newm);
            float rs = p0 + p1;
#pragma unroll
            for (int msk = 1; msk < 16; msk <<= 1)
                rs += __shfl_xor(rs, msk, 32);
            lrow[r] = lrow[r] * alpha + rs;
#pragma unroll
            for (int nt = 0; nt < 4; nt++) oacc[nt][r] *= alpha;
            const int m = r + 8 * half;
            Pl[m * 32 + ln]      = f2bf(p0);   // kv tile 0 -> k = ln
            Pl[m * 32 + 16 + ln] = f2bf(p1);   // kv tile 1 -> k = 16 + ln
        }

        // ---- P (16x32) A-fragment from LDS ----
        Frag16 pf;
        pf.q[0] = *(const uint4*)(Pl + ln * 32 + half * 8);
        pf.q[1] = *(const uint4*)(Pl + ln * 32 + 16 + half * 8);

        // ---- O += P * V ; V^T gives contiguous kv per lane ----
#pragma unroll
        for (int nt = 0; nt < 4; nt++) {
            const int d = nt * 16 + ln;
            const unsigned short* vp =
                Vt + ((size_t)(b * H_ + h) * DH_ + d) * S_ + kv0 + half * 8;
            Frag16 vf;
            vf.q[0] = *(const uint4*)(vp);
            vf.q[1] = *(const uint4*)(vp + 16);
            oacc[nt] = wmma_bf16(pf.v, vf.v, oacc[nt]);
        }
    }

    // ---- normalize and store O (bf16, [B,S,D] with head offset) ----
#pragma unroll
    for (int nt = 0; nt < 4; nt++) {
#pragma unroll
        for (int r = 0; r < 8; r++) {
            const float o = oacc[nt][r] / lrow[r];
            const size_t row = (size_t)(b * S_ + q_base + r + 8 * half);
            Ob[row * D_ + h * DH_ + nt * 16 + ln] = f2bf(o);
        }
    }
}

// ---------------- host-side orchestration ----------------
extern "C" void kernel_launch(void* const* d_in, const int* in_sizes, int n_in,
                              void* d_out, int out_size, void* d_ws, size_t ws_size,
                              hipStream_t stream) {
    (void)in_sizes; (void)n_in; (void)out_size; (void)ws_size;
    const float* X  = (const float*)d_in[0];
    const float* Wq = (const float*)d_in[1];
    const float* bq = (const float*)d_in[2];
    const float* Wk = (const float*)d_in[3];
    const float* bk = (const float*)d_in[4];
    const float* Wv = (const float*)d_in[5];
    const float* bv = (const float*)d_in[6];
    const float* Wo = (const float*)d_in[7];
    const float* bo = (const float*)d_in[8];
    const int* causal = (const int*)d_in[9];
    float* out = (float*)d_out;

    char* p = (char*)d_ws;
    auto take = [&](size_t bytes) -> char* {
        char* r = p; p += (bytes + 255) & ~(size_t)255; return r;
    };
    unsigned short* Xb  = (unsigned short*)take((size_t)M_ * D_ * 2);   // 16 MB
    unsigned short* WqT = (unsigned short*)take((size_t)D_ * D_ * 2);   //  2 MB
    unsigned short* WkT = (unsigned short*)take((size_t)D_ * D_ * 2);
    unsigned short* WvT = (unsigned short*)take((size_t)D_ * D_ * 2);
    unsigned short* WoT = (unsigned short*)take((size_t)D_ * D_ * 2);
    unsigned short* Qb  = (unsigned short*)take((size_t)M_ * D_ * 2);   // 16 MB
    unsigned short* Kb  = (unsigned short*)take((size_t)M_ * D_ * 2);   // 16 MB
    unsigned short* Vt  = (unsigned short*)take((size_t)M_ * D_ * 2);   // 16 MB [B,H,dh,S]
    unsigned short* Ob  = (unsigned short*)take((size_t)M_ * D_ * 2);   // 16 MB

    // 1) X -> bf16
    cvt_f32_bf16_kernel<<<(M_ * D_) / (256 * 4), 256, 0, stream>>>(X, Xb, M_ * D_);
    // 2) weights -> bf16 transposed
    dim3 tg(D_ / 32, D_ / 32);
    transpose_cvt_kernel<<<tg, 256, 0, stream>>>(Wq, WqT);
    transpose_cvt_kernel<<<tg, 256, 0, stream>>>(Wk, WkT);
    transpose_cvt_kernel<<<tg, 256, 0, stream>>>(Wv, WvT);
    transpose_cvt_kernel<<<tg, 256, 0, stream>>>(Wo, WoT);
    // 3) QKV projections (V scattered straight to transposed layout)
    dim3 gg(M_ / 256, D_ / 64);
    gemm_bf16_kernel<0><<<gg, 256, 0, stream>>>(Xb, WqT, bq, Qb);
    gemm_bf16_kernel<0><<<gg, 256, 0, stream>>>(Xb, WkT, bk, Kb);
    gemm_bf16_kernel<1><<<gg, 256, 0, stream>>>(Xb, WvT, bv, Vt);
    // 4) flash attention
    attn_kernel<<<B_ * H_ * (S_ / 128), 256, 0, stream>>>(Qb, Kb, Vt, Ob, causal);
    // 5) output projection (f32 out)
    gemm_bf16_kernel<2><<<gg, 256, 0, stream>>>(Ob, WoT, bo, out);
}
